// StateEstimator_57732950393557
// MI455X (gfx1250) — matrix operand order repaired and checked by reading, not compiled, not run on hardware
//
#include <hip/hip_runtime.h>
#include <math.h>

typedef __bf16 bf16;
typedef __attribute__((ext_vector_type(8)))  __bf16 bf16x8;
typedef __attribute__((ext_vector_type(16))) __bf16 bf16x16;
typedef __attribute__((ext_vector_type(8)))  float  f32x8;

#define WMMA_BF16(a, b, c) \
  __builtin_amdgcn_wmma_f32_16x16x32_bf16(false, (a), false, (b), (short)0, (c), false, false)

#define F8Z {0.f,0.f,0.f,0.f,0.f,0.f,0.f,0.f}

#define CAT16(lo, hi) \
  __builtin_shufflevector((lo), (hi), 0, 1, 2, 3, 4, 5, 6, 7, 8, 9, 10, 11, 12, 13, 14, 15)

// ---------------- WMMA fragment helpers (ISA 7.12.2 layouts) ----------------

// A fragment: 16x32 bf16 from row-major source (lda elements), K chunk at k0.
__device__ __forceinline__ bf16x16 a_frag(const bf16* A, int lda, int lane, int k0) {
  int m = lane & 15;
  int kb = (lane >> 4) << 3;                 // 0 or 8
  const bf16* base = A + (long)m * lda + k0 + kb;
  bf16x8 lo = *(const bf16x8*)base;          // K: kb .. kb+7
  bf16x8 hi = *(const bf16x8*)(base + 16);   // K: 16+kb .. 16+kb+7
  return CAT16(lo, hi);
}

// B fragment where element (k,n) lives at S[(n0+n)*ld + k0+k]  (S = B^T,
// contraction contiguous). lanes 0-15: K 0..7,16..23; lanes 16-31: K 8..15,24..31.
__device__ __forceinline__ bf16x16 b_frag_t(const bf16* S, int ld, int lane, int k0, int n0) {
  int n = (lane & 15) + n0;
  int kb = (lane >> 4) << 3;                 // 0 or 8
  const bf16* base = S + (long)n * ld + k0 + kb;
  bf16x8 lo = *(const bf16x8*)base;
  bf16x8 hi = *(const bf16x8*)(base + 16);
  return CAT16(lo, hi);
}

__device__ __forceinline__ float gelu_exact(float x) {
  return 0.5f * x * (1.f + erff(x * 0.70710678118654752f));
}

// ---------------- small utility kernels ----------------

__global__ void cvt_bf16_kernel(const float* __restrict__ src, bf16* __restrict__ dst, int n) {
  int i = blockIdx.x * 256 + threadIdx.x;
  if (i < n) dst[i] = (bf16)src[i];
}

// dst (N x K bf16, row-major) = transpose of src (K x N fp32, row-major)
__global__ void cvt_bf16_t_kernel(const float* __restrict__ src, bf16* __restrict__ dst,
                                  int K, int N) {
  int i = blockIdx.x * 256 + threadIdx.x;
  if (i < K * N) {
    int k = i / N, n = i - k * N;
    dst[(long)n * K + k] = (bf16)src[i];
  }
}

__global__ void init_slots_kernel(const float* __restrict__ mu, const float* __restrict__ ls,
                                  const float* __restrict__ noise, float* __restrict__ slots, int n) {
  int i = blockIdx.x * 256 + threadIdx.x;
  if (i < n) {
    int j = i & (8 * 256 - 1);
    slots[i] = mu[j] + __expf(ls[j]) * noise[i];
  }
}

// ---------------- kernel 1: inputs = LN(patch @ Wp + bp) -> bf16 ----------------
// 512 thr = 16 waves: rowgrp = wv>>1 (8 x 16 rows), colgrp = wv&1 (2 x 128 cols).
// Block tile: 128 rows x 256 cols. WpT (256 x 192) staged in 32-K LDS slabs (2x16KB).
// Per-wave accs: 4 n-tiles x 2 = 8 f32x8 (64 VGPRs) -> no spills.
__global__ __launch_bounds__(512, 1)
void proj_ln_big_kernel(const float* __restrict__ patch, const bf16* __restrict__ WpT,
                        const float* __restrict__ bp, const float* __restrict__ g_in,
                        const float* __restrict__ b_in, bf16* __restrict__ inputs_bf) {
  extern __shared__ char smem[];
  bf16*  wslab = (bf16*)smem;                    // 2 x 256 x 32
  float* redS  = (float*)(wslab + 2 * 256 * 32); // 128 rows x 2 colgrps
  float* redSS = redS + 256;
  const int tid = threadIdx.x, lane = tid & 31, wv = tid >> 5;
  const int rowgrp = wv >> 1, colgrp = wv & 1;
  const long row0 = (long)blockIdx.x * 128;
  const int m = lane & 15, kbA = (lane >> 4) << 3;
  const float* arow = patch + (row0 + rowgrp * 16 + m) * 192;

  f32x8 acc[4][2];
  {
    f32x8 z = F8Z;
#pragma unroll
    for (int nt = 0; nt < 4; ++nt) { acc[nt][0] = z; acc[nt][1] = z; }
  }

  auto stage = [&](int buf, int k0) {
    int row = tid >> 1, part = (tid & 1) * 16;
    const bf16* src = WpT + (long)row * 192 + k0 + part;
    bf16* dst = wslab + buf * 256 * 32 + row * 32 + part;
    *(bf16x8*)(dst)     = *(const bf16x8*)(src);
    *(bf16x8*)(dst + 8) = *(const bf16x8*)(src + 8);
  };

  stage(0, 0);
  __syncthreads();
  for (int kc = 0; kc < 6; ++kc) {
    int buf = kc & 1;
    if (kc < 5) stage(buf ^ 1, (kc + 1) * 32);
    int k0 = kc * 32;
    bf16x16 a;
#pragma unroll
    for (int i = 0; i < 8; ++i) {
      a[i]     = (bf16)arow[k0 + kbA + i];
      a[8 + i] = (bf16)arow[k0 + 16 + kbA + i];
    }
    const bf16* wb = wslab + buf * 256 * 32;
#pragma unroll
    for (int nt = 0; nt < 4; ++nt) {
      int nn = colgrp * 128 + nt * 32;
      bf16x16 b0 = b_frag_t(wb, 32, lane, 0, nn);
      bf16x16 b1 = b_frag_t(wb, 32, lane, 0, nn + 16);
      acc[nt][0] = WMMA_BF16(a, b0, acc[nt][0]);
      acc[nt][1] = WMMA_BF16(a, b1, acc[nt][1]);
    }
    __syncthreads();
  }

  // epilogue: +bias, LayerNorm (shfl partials + 2-way LDS combine), -> bf16
  const int n = lane & 15, mb = (lane >> 4) << 3;
  float gw[8], bw[8];
#pragma unroll
  for (int nt = 0; nt < 4; ++nt)
#pragma unroll
    for (int h = 0; h < 2; ++h) {
      int col = colgrp * 128 + nt * 32 + h * 16 + n;
      float bias = bp[col];
      gw[nt * 2 + h] = g_in[col];
      bw[nt * 2 + h] = b_in[col];
#pragma unroll
      for (int r = 0; r < 8; ++r) acc[nt][h][r] += bias;
    }
#pragma unroll
  for (int r = 0; r < 8; ++r) {
    float s = 0.f, ss = 0.f;
#pragma unroll
    for (int nt = 0; nt < 4; ++nt)
#pragma unroll
      for (int h = 0; h < 2; ++h) { float v = acc[nt][h][r]; s += v; ss += v * v; }
#pragma unroll
    for (int msk = 1; msk < 16; msk <<= 1) {
      s  += __shfl_xor(s, msk, 32);
      ss += __shfl_xor(ss, msk, 32);
    }
    if ((lane & 15) == 0) {
      int ridx = rowgrp * 16 + mb + r;
      redS[ridx * 2 + colgrp]  = s;
      redSS[ridx * 2 + colgrp] = ss;
    }
  }
  __syncthreads();
#pragma unroll
  for (int r = 0; r < 8; ++r) {
    int ridx = rowgrp * 16 + mb + r;
    float s  = redS[ridx * 2 + 0] + redS[ridx * 2 + 1];
    float ss = redSS[ridx * 2 + 0] + redSS[ridx * 2 + 1];
    float mu = s * (1.f / 256.f);
    float rs = rsqrtf(ss * (1.f / 256.f) - mu * mu + 1e-5f);
    long obase = (row0 + ridx) * 256;
#pragma unroll
    for (int nt = 0; nt < 4; ++nt)
#pragma unroll
      for (int h = 0; h < 2; ++h) {
        int col = colgrp * 128 + nt * 32 + h * 16 + n;
        float v = (acc[nt][h][r] - mu) * rs * gw[nt * 2 + h] + bw[nt * 2 + h];
        inputs_bf[obase + col] = (bf16)v;
      }
  }
}

// ---------------- kernel 2: big staged GEMM, K=256 ----------------
// grid = (M/128, Ntot/256); block = 512 thr (16 waves; wave = 16 rows x 128 cols)
// OUT_MODE 0: bf16 row-major + bias
// OUT_MODE 1: bf16 per-batch transposed (C[b][col][token]) + bias (Ntot == 256)
// OUT_MODE 2: f32 row-major (ld = Ntot) + bias
template <int OUT_MODE>
__global__ __launch_bounds__(512, 1)
void gemm_big_kernel(const bf16* __restrict__ A, const bf16* __restrict__ WT,
                     const float* __restrict__ bias, void* __restrict__ Cout, int Ntot) {
  extern __shared__ char smem[];
  bf16* wslab = (bf16*)smem;                 // 2 x 256 x 32
  const int tid = threadIdx.x, lane = tid & 31, wv = tid >> 5;
  const int rowgrp = wv >> 1, colgrp = wv & 1;
  const long row0 = (long)blockIdx.x * 128;
  const int nbase = blockIdx.y * 256;
  const bf16* Arow = A + (row0 + rowgrp * 16) * 256;

  f32x8 acc[4][2];
  {
    f32x8 z = F8Z;
#pragma unroll
    for (int nt = 0; nt < 4; ++nt) { acc[nt][0] = z; acc[nt][1] = z; }
  }

  auto stage = [&](int buf, int k0) {
    int row = tid >> 1, part = (tid & 1) * 16;
    const bf16* src = WT + (long)(nbase + row) * 256 + k0 + part;
    bf16* dst = wslab + buf * 256 * 32 + row * 32 + part;
    *(bf16x8*)(dst)     = *(const bf16x8*)(src);
    *(bf16x8*)(dst + 8) = *(const bf16x8*)(src + 8);
  };

  stage(0, 0);
  __syncthreads();
  for (int kc = 0; kc < 8; ++kc) {
    int buf = kc & 1;
    if (kc < 7) stage(buf ^ 1, (kc + 1) * 32);
    bf16x16 a = a_frag(Arow, 256, lane, kc * 32);
    const bf16* wb = wslab + buf * 256 * 32;
#pragma unroll
    for (int nt = 0; nt < 4; ++nt) {
      int nn = colgrp * 128 + nt * 32;
      bf16x16 b0 = b_frag_t(wb, 32, lane, 0, nn);
      bf16x16 b1 = b_frag_t(wb, 32, lane, 0, nn + 16);
      acc[nt][0] = WMMA_BF16(a, b0, acc[nt][0]);
      acc[nt][1] = WMMA_BF16(a, b1, acc[nt][1]);
    }
    __syncthreads();
  }

  const int n = lane & 15, mb = (lane >> 4) << 3;
  if (OUT_MODE == 1) {
    // rows of this block are 128 consecutive tokens within one 256-token batch
    long bidx = row0 >> 8;
    int  tl0  = (int)(row0 & 255) + rowgrp * 16 + mb;
    bf16* Cb = (bf16*)Cout + bidx * 256 * 256;
#pragma unroll
    for (int nt = 0; nt < 4; ++nt)
#pragma unroll
      for (int h = 0; h < 2; ++h) {
        int col = colgrp * 128 + nt * 32 + h * 16 + n;
        float bb = bias[col];
        bf16x8 pk;
#pragma unroll
        for (int r = 0; r < 8; ++r) pk[r] = (bf16)(acc[nt][h][r] + bb);
        *(bf16x8*)(Cb + (long)col * 256 + tl0) = pk;
      }
  } else if (OUT_MODE == 0) {
    bf16* C = (bf16*)Cout;
#pragma unroll
    for (int nt = 0; nt < 4; ++nt)
#pragma unroll
      for (int h = 0; h < 2; ++h) {
        int col = nbase + colgrp * 128 + nt * 32 + h * 16 + n;
        float bb = bias[col];
#pragma unroll
        for (int r = 0; r < 8; ++r)
          C[(row0 + rowgrp * 16 + mb + r) * (long)Ntot + col] = (bf16)(acc[nt][h][r] + bb);
      }
  } else {
    float* C = (float*)Cout;
#pragma unroll
    for (int nt = 0; nt < 4; ++nt)
#pragma unroll
      for (int h = 0; h < 2; ++h) {
        int col = nbase + colgrp * 128 + nt * 32 + h * 16 + n;
        float bb = bias[col];
#pragma unroll
        for (int r = 0; r < 8; ++r)
          C[(row0 + rowgrp * 16 + mb + r) * (long)Ntot + col] = acc[nt][h][r] + bb;
      }
  }
}

// ---------------- kernel 3: GRU gates elementwise ----------------
__global__ void gru_gates_kernel(const float* __restrict__ gi, const float* __restrict__ gh,
                                 float* __restrict__ slots) {
  const long row = blockIdx.x;
  const int j = threadIdx.x;
  const float* gir = gi + row * 768;
  const float* ghr = gh + row * 768;
  float ir  = gir[j], iz = gir[256 + j], inn = gir[512 + j];
  float hr  = ghr[j], hz = ghr[256 + j], hn  = ghr[512 + j];
  float rg = 1.f / (1.f + __expf(-(ir + hr)));
  float zg = 1.f / (1.f + __expf(-(iz + hz)));
  float ng = tanhf(inn + rg * hn);
  float h  = slots[row * 256 + j];
  slots[row * 256 + j] = (1.f - zg) * ng + zg * h;
}

// ---------------- kernel 4: s = LN(slots); q = s @ Wq + bq -> bf16 ----------------
__global__ void ln_gemm_q_kernel(const float* __restrict__ slots, const float* __restrict__ g,
                                 const float* __restrict__ beta, const bf16* __restrict__ WqT,
                                 const float* __restrict__ bq, bf16* __restrict__ qbf) {
  extern __shared__ char smem[];
  bf16* at   = (bf16*)smem;                 // 16 x 256 bf16 (LN output = A tile)
  float* red = (float*)(at + 16 * 256);
  __shared__ float s_mu[16], s_rs[16];
  const int tid = threadIdx.x, lane = tid & 31, wv = tid >> 5;
  const long row0 = (long)blockIdx.x * 16;

  int r = tid >> 4, sg = (tid & 15) * 16;
  const float* srow = slots + (row0 + r) * 256;
  float s = 0.f, ss = 0.f;
#pragma unroll
  for (int i = 0; i < 16; ++i) { float v = srow[sg + i]; s += v; ss += v * v; }
  red[tid] = s; red[256 + tid] = ss;
  __syncthreads();
  if (tid < 16) {
    float S = 0.f, SS = 0.f;
    for (int i = 0; i < 16; ++i) { S += red[tid * 16 + i]; SS += red[256 + tid * 16 + i]; }
    float m_ = S * (1.f / 256.f);
    s_mu[tid] = m_;
    s_rs[tid] = rsqrtf(SS * (1.f / 256.f) - m_ * m_ + 1e-5f);
  }
  __syncthreads();
  {
    float m_ = s_mu[r], rr = s_rs[r];
#pragma unroll
    for (int i = 0; i < 16; ++i)
      at[r * 256 + sg + i] = (bf16)((srow[sg + i] - m_) * rr * g[sg + i] + beta[sg + i]);
  }
  __syncthreads();
  const int n0 = wv * 32;
  f32x8 acc0 = F8Z, acc1 = F8Z;
#pragma unroll
  for (int kc = 0; kc < 8; ++kc) {
    int k0 = kc * 32;
    bf16x16 a  = a_frag(at, 256, lane, k0);
    bf16x16 b0 = b_frag_t(WqT, 256, lane, k0, n0);
    bf16x16 b1 = b_frag_t(WqT, 256, lane, k0, n0 + 16);
    acc0 = WMMA_BF16(a, b0, acc0);
    acc1 = WMMA_BF16(a, b1, acc1);
  }
  int n = lane & 15, mb = (lane >> 4) << 3;
#pragma unroll
  for (int r2 = 0; r2 < 8; ++r2) {
    long base = (row0 + mb + r2) * 256;
    qbf[base + n0 + n]      = (bf16)(acc0[r2] + bq[n0 + n]);
    qbf[base + n0 + 16 + n] = (bf16)(acc1[r2] + bq[n0 + 16 + n]);
  }
}

// ---------------- kernel 5: attention + weighted update, one block per batch ----------------
__global__ void attn_update_kernel(const bf16* __restrict__ qbf, const bf16* __restrict__ kbf,
                                   const bf16* __restrict__ vTbf, bf16* __restrict__ updbf) {
  extern __shared__ char smem[];
  bf16* qa    = (bf16*)smem;                 // 16 x 256 (rows 8..15 zero pad)
  float* attf = (float*)(qa + 16 * 256);     // 16 x 256 logits
  bf16* atb   = (bf16*)(attf + 16 * 256);    // 16 x 256 probs
  const int tid = threadIdx.x, lane = tid & 31, wv = tid >> 5;
  const long b = blockIdx.x;
  const bf16* kb_ = kbf + b * 256 * 256;
  const bf16* vb_ = vTbf + b * 256 * 256;

  for (int i = tid; i < 16 * 256; i += 256) {
    int rr = i >> 8, cc = i & 255;
    qa[i] = (rr < 8) ? qbf[(b * 8 + rr) * 256 + cc] : (bf16)0.f;
  }
  __syncthreads();

  {
    const int n0 = wv * 32;
    f32x8 acc0 = F8Z, acc1 = F8Z;
#pragma unroll
    for (int kc = 0; kc < 8; ++kc) {
      int k0 = kc * 32;
      bf16x16 a  = a_frag(qa, 256, lane, k0);
      bf16x16 b0 = b_frag_t(kb_, 256, lane, k0, n0);
      bf16x16 b1 = b_frag_t(kb_, 256, lane, k0, n0 + 16);
      acc0 = WMMA_BF16(a, b0, acc0);
      acc1 = WMMA_BF16(a, b1, acc1);
    }
    int n = lane & 15, mb = (lane >> 4) << 3;
#pragma unroll
    for (int r = 0; r < 8; ++r) {
      attf[(mb + r) * 256 + n0 + n]      = acc0[r] * 0.0625f;
      attf[(mb + r) * 256 + n0 + 16 + n] = acc1[r] * 0.0625f;
    }
  }
  __syncthreads();

  {
    int n = tid;
    float mx = -1e30f;
#pragma unroll
    for (int r = 0; r < 8; ++r) mx = fmaxf(mx, attf[r * 256 + n]);
    float e[8], sm = 0.f;
#pragma unroll
    for (int r = 0; r < 8; ++r) { e[r] = __expf(attf[r * 256 + n] - mx); sm += e[r]; }
    float inv = 1.f / sm;
#pragma unroll
    for (int r = 0; r < 8; ++r) atb[r * 256 + n] = (bf16)(e[r] * inv);
#pragma unroll
    for (int r = 8; r < 16; ++r) atb[r * 256 + n] = (bf16)0.f;
  }
  __syncthreads();

  {
    const int d0 = wv * 32;
    f32x8 acc0 = F8Z, acc1 = F8Z;
#pragma unroll
    for (int kc = 0; kc < 8; ++kc) {
      int k0 = kc * 32;
      bf16x16 a  = a_frag(atb, 256, lane, k0);
      bf16x16 b0 = b_frag_t(vb_, 256, lane, k0, d0);      // vT: (d, token)
      bf16x16 b1 = b_frag_t(vb_, 256, lane, k0, d0 + 16);
      acc0 = WMMA_BF16(a, b0, acc0);
      acc1 = WMMA_BF16(a, b1, acc1);
    }
    int n = lane & 15, mb = (lane >> 4) << 3;
#pragma unroll
    for (int r = 0; r < 8; ++r) {
      int row = mb + r;
      if (row < 8) {
        long base = (b * 8 + row) * 256;
        updbf[base + d0 + n]      = (bf16)acc0[r];
        updbf[base + d0 + 16 + n] = (bf16)acc1[r];
      }
    }
  }
}

// ---------------- kernel 6: fused residual MLP ----------------
__global__ void mlp_kernel(float* __restrict__ slots, const float* __restrict__ g,
                           const float* __restrict__ beta, const bf16* __restrict__ W1T,
                           const float* __restrict__ b1, const bf16* __restrict__ W2T,
                           const float* __restrict__ b2) {
  extern __shared__ char smem[];
  float* xrow = (float*)smem;               // 16 x 256 residual
  bf16*  at   = (bf16*)(xrow + 16 * 256);   // 16 x 256 LN output
  bf16*  hid  = (bf16*)(at + 16 * 256);     // 16 x 512 gelu hidden
  float* red  = (float*)(hid + 16 * 512);
  __shared__ float s_mu[16], s_rs[16];
  const int tid = threadIdx.x, lane = tid & 31, wv = tid >> 5;
  const long row0 = (long)blockIdx.x * 16;

  int r = tid >> 4, sg = (tid & 15) * 16;
  float s = 0.f, ss = 0.f;
#pragma unroll
  for (int i = 0; i < 16; ++i) {
    float v = slots[(row0 + r) * 256 + sg + i];
    xrow[r * 256 + sg + i] = v;
    s += v; ss += v * v;
  }
  red[tid] = s; red[256 + tid] = ss;
  __syncthreads();
  if (tid < 16) {
    float S = 0.f, SS = 0.f;
    for (int i = 0; i < 16; ++i) { S += red[tid * 16 + i]; SS += red[256 + tid * 16 + i]; }
    float m_ = S * (1.f / 256.f);
    s_mu[tid] = m_;
    s_rs[tid] = rsqrtf(SS * (1.f / 256.f) - m_ * m_ + 1e-5f);
  }
  __syncthreads();
  {
    float m_ = s_mu[r], rr = s_rs[r];
#pragma unroll
    for (int i = 0; i < 16; ++i)
      at[r * 256 + sg + i] = (bf16)((xrow[r * 256 + sg + i] - m_) * rr * g[sg + i] + beta[sg + i]);
  }
  __syncthreads();

#pragma unroll
  for (int p = 0; p < 2; ++p) {
    const int n0 = wv * 32 + p * 256;
    f32x8 a0 = F8Z, a1 = F8Z;
#pragma unroll
    for (int kc = 0; kc < 8; ++kc) {
      int k0 = kc * 32;
      bf16x16 a  = a_frag(at, 256, lane, k0);
      bf16x16 c0 = b_frag_t(W1T, 256, lane, k0, n0);
      bf16x16 c1 = b_frag_t(W1T, 256, lane, k0, n0 + 16);
      a0 = WMMA_BF16(a, c0, a0);
      a1 = WMMA_BF16(a, c1, a1);
    }
    int n = lane & 15, mb = (lane >> 4) << 3;
#pragma unroll
    for (int q = 0; q < 8; ++q) {
      float v0 = a0[q] + b1[n0 + n];
      float v1 = a1[q] + b1[n0 + 16 + n];
      hid[(mb + q) * 512 + n0 + n]      = (bf16)gelu_exact(v0);
      hid[(mb + q) * 512 + n0 + 16 + n] = (bf16)gelu_exact(v1);
    }
  }
  __syncthreads();

  {
    const int n0 = wv * 32;
    f32x8 a0 = F8Z, a1 = F8Z;
#pragma unroll
    for (int kc = 0; kc < 16; ++kc) {
      int k0 = kc * 32;
      bf16x16 a  = a_frag(hid, 512, lane, k0);
      bf16x16 c0 = b_frag_t(W2T, 512, lane, k0, n0);
      bf16x16 c1 = b_frag_t(W2T, 512, lane, k0, n0 + 16);
      a0 = WMMA_BF16(a, c0, a0);
      a1 = WMMA_BF16(a, c1, a1);
    }
    int n = lane & 15, mb = (lane >> 4) << 3;
#pragma unroll
    for (int q = 0; q < 8; ++q) {
      long base = (row0 + mb + q) * 256;
      slots[base + n0 + n]      = xrow[(mb + q) * 256 + n0 + n] + a0[q] + b2[n0 + n];
      slots[base + n0 + 16 + n] = xrow[(mb + q) * 256 + n0 + 16 + n] + a1[q] + b2[n0 + 16 + n];
    }
  }
}

// ---------------- kernel 7: output heads ----------------
__global__ void heads_kernel(const float* __restrict__ slots,
                             const bf16* Wg1T, const float* bg1, const float* Wg2, const float* bg2,
                             const bf16* Wm1T, const float* bm1, const float* Wm2, const float* bm2,
                             const bf16* Wu1T, const float* bu1, const float* Wu2, const float* bu2,
                             float* __restrict__ out_slots, float* __restrict__ out_qv,
                             float* __restrict__ out_z, float* __restrict__ out_sig) {
  extern __shared__ char smem[];
  bf16*  at = (bf16*)smem;              // 16 x 256
  float* t1 = (float*)(at + 16 * 256);  // 16 x 256
  const int tid = threadIdx.x, lane = tid & 31, wv = tid >> 5;
  const long row0 = (long)blockIdx.x * 16;

  for (int i = tid; i < 16 * 256; i += 256) {
    float v = slots[row0 * 256 + i];
    at[i] = (bf16)v;
    out_slots[row0 * 256 + i] = v;
  }
  __syncthreads();

  for (int hsel = 0; hsel < 3; ++hsel) {
    const bf16*  W1p = (hsel == 0) ? Wg1T : ((hsel == 1) ? Wm1T : Wu1T);
    const float* b1p = (hsel == 0) ? bg1 : ((hsel == 1) ? bm1 : bu1);
    const float* W2p = (hsel == 0) ? Wg2 : ((hsel == 1) ? Wm2 : Wu2);
    const float* b2p = (hsel == 0) ? bg2 : ((hsel == 1) ? bm2 : bu2);
    const int ow = (hsel == 0) ? 4 : 8;
    float* outp = (hsel == 0) ? out_qv : ((hsel == 1) ? out_z : out_sig);

    const int n0 = wv * 32;
    f32x8 a0 = F8Z, a1 = F8Z;
#pragma unroll
    for (int kc = 0; kc < 8; ++kc) {
      int k0 = kc * 32;
      bf16x16 a  = a_frag(at, 256, lane, k0);
      bf16x16 c0 = b_frag_t(W1p, 256, lane, k0, n0);
      bf16x16 c1 = b_frag_t(W1p, 256, lane, k0, n0 + 16);
      a0 = WMMA_BF16(a, c0, a0);
      a1 = WMMA_BF16(a, c1, a1);
    }
    int n = lane & 15, mb = (lane >> 4) << 3;
#pragma unroll
    for (int q = 0; q < 8; ++q) {
      float v0 = a0[q] + b1p[n0 + n];
      float v1 = a1[q] + b1p[n0 + 16 + n];
      t1[(mb + q) * 256 + n0 + n]      = gelu_exact(v0);
      t1[(mb + q) * 256 + n0 + 16 + n] = gelu_exact(v1);
    }
    __syncthreads();
    if (tid < 16 * ow) {
      int rr = tid / ow, c = tid % ow;
      float acc = b2p[c];
      for (int d = 0; d < 256; ++d) acc += t1[rr * 256 + d] * W2p[d * ow + c];
      if (hsel == 2) acc = (acc > 20.f) ? acc : log1pf(__expf(acc));
      outp[(row0 + rr) * ow + c] = acc;
    }
    __syncthreads();
  }
}

// ---------------- host launcher ----------------
extern "C" void kernel_launch(void* const* d_in, const int* in_sizes, int n_in,
                              void* d_out, int out_size, void* d_ws, size_t ws_size,
                              hipStream_t stream) {
  (void)in_sizes; (void)n_in; (void)out_size; (void)ws_size;
  const float* patch = (const float*)d_in[0];
  const float* noise = (const float*)d_in[1];
  const float* s_mu  = (const float*)d_in[2];
  const float* s_ls  = (const float*)d_in[3];
  const float* Wp  = (const float*)d_in[4];  const float* bp  = (const float*)d_in[5];
  const float* g_in= (const float*)d_in[6];  const float* b_in= (const float*)d_in[7];
  const float* Wq  = (const float*)d_in[8];  const float* bq  = (const float*)d_in[9];
  const float* Wk  = (const float*)d_in[10]; const float* bk  = (const float*)d_in[11];
  const float* Wv  = (const float*)d_in[12]; const float* bv  = (const float*)d_in[13];
  const float* Wih = (const float*)d_in[14]; const float* bih = (const float*)d_in[15];
  const float* Whh = (const float*)d_in[16]; const float* bhh = (const float*)d_in[17];
  const float* g_s = (const float*)d_in[18]; const float* b_s = (const float*)d_in[19];
  const float* W1  = (const float*)d_in[20]; const float* b1  = (const float*)d_in[21];
  const float* W2  = (const float*)d_in[22]; const float* b2  = (const float*)d_in[23];
  const float* g_m = (const float*)d_in[24]; const float* b_m = (const float*)d_in[25];
  const float* Wg1 = (const float*)d_in[26]; const float* bg1 = (const float*)d_in[27];
  const float* Wg2 = (const float*)d_in[28]; const float* bg2 = (const float*)d_in[29];
  const float* Wm1 = (const float*)d_in[30]; const float* bm1 = (const float*)d_in[31];
  const float* Wm2 = (const float*)d_in[32]; const float* bm2 = (const float*)d_in[33];
  const float* Wu1 = (const float*)d_in[34]; const float* bu1 = (const float*)d_in[35];
  const float* Wu2 = (const float*)d_in[36]; const float* bu2 = (const float*)d_in[37];

  const long BN = 1024L * 256;        // token rows
  const long BK = 1024L * 8;          // slot rows

  char* w = (char*)d_ws;
  auto alloc = [&](size_t bytes) { char* p = w; w += (bytes + 255) & ~(size_t)255; return p; };

  bf16* WpT_b  = (bf16*)alloc(192 * 256 * 2);   // (256 x 192)
  bf16* WqT_b  = (bf16*)alloc(256 * 256 * 2);
  bf16* WkT_b  = (bf16*)alloc(256 * 256 * 2);
  bf16* WvT_b  = (bf16*)alloc(256 * 256 * 2);
  bf16* Wih_b  = (bf16*)alloc(768 * 256 * 2);   // native layout (K-contiguous rows)
  bf16* Whh_b  = (bf16*)alloc(768 * 256 * 2);
  bf16* W1T_b  = (bf16*)alloc(256 * 512 * 2);   // (512 x 256)
  bf16* W2T_b  = (bf16*)alloc(512 * 256 * 2);   // (256 x 512)
  bf16* Wg1T_b = (bf16*)alloc(256 * 256 * 2);
  bf16* Wm1T_b = (bf16*)alloc(256 * 256 * 2);
  bf16* Wu1T_b = (bf16*)alloc(256 * 256 * 2);
  bf16* inp_b  = (bf16*)alloc((size_t)BN * 256 * 2);
  bf16* k_b    = (bf16*)alloc((size_t)BN * 256 * 2);
  bf16* vT_b   = (bf16*)alloc((size_t)BN * 256 * 2);  // per-batch (d, token)
  bf16* q_b    = (bf16*)alloc((size_t)BK * 256 * 2);
  bf16* upd_b  = (bf16*)alloc((size_t)BK * 256 * 2);
  bf16* h_b    = (bf16*)alloc((size_t)BK * 256 * 2);  // slots as bf16 (GRU input)
  float* slots = (float*)alloc((size_t)BK * 256 * 4);
  float* gi_f  = (float*)alloc((size_t)BK * 768 * 4);
  float* gh_f  = (float*)alloc((size_t)BK * 768 * 4);

  auto cvt = [&](const float* src, bf16* dst, int n) {
    cvt_bf16_kernel<<<(n + 255) / 256, 256, 0, stream>>>(src, dst, n);
  };
  auto cvtT = [&](const float* src, bf16* dst, int K, int N) {
    cvt_bf16_t_kernel<<<(K * N + 255) / 256, 256, 0, stream>>>(src, dst, K, N);
  };
  cvtT(Wp, WpT_b, 192, 256);
  cvtT(Wq, WqT_b, 256, 256);
  cvtT(Wk, WkT_b, 256, 256);
  cvtT(Wv, WvT_b, 256, 256);
  cvt (Wih, Wih_b, 768 * 256);
  cvt (Whh, Whh_b, 768 * 256);
  cvtT(W1, W1T_b, 256, 512);
  cvtT(W2, W2T_b, 512, 256);
  cvtT(Wg1, Wg1T_b, 256, 256);
  cvtT(Wm1, Wm1T_b, 256, 256);
  cvtT(Wu1, Wu1T_b, 256, 256);

  init_slots_kernel<<<(int)((BK * 256 + 255) / 256), 256, 0, stream>>>(
      s_mu, s_ls, noise, slots, (int)(BK * 256));

  const size_t slab2  = 2 * 256 * 32 * sizeof(bf16);       // 32 KB W slabs
  const size_t slabLN = slab2 + 512 * sizeof(float);       // + LN partials
  proj_ln_big_kernel<<<(int)(BN / 128), 512, slabLN, stream>>>(
      patch, WpT_b, bp, g_in, b_in, inp_b);

  // k and v are loop-invariant in the reference -> compute once
  gemm_big_kernel<0><<<dim3((int)(BN / 128), 1), 512, slab2, stream>>>(
      inp_b, WkT_b, bk, k_b, 256);
  gemm_big_kernel<1><<<dim3((int)(BN / 128), 1), 512, slab2, stream>>>(
      inp_b, WvT_b, bv, vT_b, 256);

  for (int it = 0; it < 3; ++it) {
    ln_gemm_q_kernel<<<(int)(BK / 16), 256, 16 * 256 * 2 + 512 * 4, stream>>>(
        slots, g_s, b_s, WqT_b, bq, q_b);
    attn_update_kernel<<<1024, 256, 32768, stream>>>(q_b, k_b, vT_b, upd_b);
    // GRU: two staged GEMMs (x@Wih^T, h@Whh^T) -> fp32 gates -> elementwise
    cvt_bf16_kernel<<<(int)(BK * 256 / 256), 256, 0, stream>>>(slots, h_b, (int)(BK * 256));
    gemm_big_kernel<2><<<dim3((int)(BK / 128), 3), 512, slab2, stream>>>(
        upd_b, Wih_b, bih, gi_f, 768);
    gemm_big_kernel<2><<<dim3((int)(BK / 128), 3), 512, slab2, stream>>>(
        h_b, Whh_b, bhh, gh_f, 768);
    gru_gates_kernel<<<(int)BK, 256, 0, stream>>>(gi_f, gh_f, slots);
    mlp_kernel<<<(int)(BK / 16), 256, 43008, stream>>>(
        slots, g_m, b_m, W1T_b, b1, W2T_b, b2);
  }

  float* out = (float*)d_out;
  float* out_slots = out;                       // 1024*8*256
  float* out_qv    = out + 2097152;             // 1024*8*4
  float* out_z     = out + 2097152 + 32768;     // 1024*8*8
  float* out_sig   = out + 2097152 + 32768 + 65536;
  heads_kernel<<<(int)(BK / 16), 256, 24576, stream>>>(
      slots, Wg1T_b, bg1, Wg2, bg2, Wm1T_b, bm1, Wm2, bm2, Wu1T_b, bu1, Wu2, bu2,
      out_slots, out_qv, out_z, out_sig);
}
